// RelRepBetweenContext_old_35957466202777
// MI455X (gfx1250) — matrix operand order repaired and checked by reading, not compiled, not run on hardware
//
#include <hip/hip_runtime.h>
#include <hip/hip_bf16.h>

// ---------------------------------------------------------------------------
// Problem constants (from the reference): B=4, K=32, S=1024, H=512
//   3H = 1536, FFN = 2304, M = B*K*K = 4096 rows
// ---------------------------------------------------------------------------
#define BB 4
#define KK 32
#define SS 1024
#define HH 512
#define H3 1536
#define FFN 2304
#define MROWS (BB * KK * KK)   // 4096

typedef __bf16 bf16_t;
typedef __attribute__((ext_vector_type(16))) __bf16 v16bf;
typedef __attribute__((ext_vector_type(8)))  __bf16 v8bf;
typedef __attribute__((ext_vector_type(8)))  float  v8f;

// ---------------------------------------------------------------------------
// Transpose + convert: in fp32 [R][C] row-major -> out bf16 [C][R] row-major
// ---------------------------------------------------------------------------
__global__ void __launch_bounds__(256)
transpose_to_bf16(const float* __restrict__ in, bf16_t* __restrict__ out,
                  int R, int C) {
    long idx = (long)blockIdx.x * blockDim.x + threadIdx.x;
    long total = (long)R * C;
    if (idx < total) {
        int r = (int)(idx / C);
        int c = (int)(idx % C);
        out[(long)c * R + r] = (bf16_t)in[idx];
    }
}

// ---------------------------------------------------------------------------
// Shared range-max precompute.
// pooled(lo, hi) depends only on lo in {ends} (K values) and hi in {starts}
// (K values) per batch. One block per (batch b, start-index q): scan tokens
// from t = starts[q]-1 downward keeping a running masked max; every time t
// crosses ends[p] (sorted descending in LDS) snapshot into ctx[b][p][q][:].
// 128 blocks total; <= ~268 MB L2 traffic vs ~4 GB for per-pair scans.
// ---------------------------------------------------------------------------
__global__ void __launch_bounds__(256)
build_ctx(const float* __restrict__ tokens,      // [B][S][H]
          const int* __restrict__ ids,           // [B][K][2]
          const unsigned char* __restrict__ msk, // [B][S]
          float* __restrict__ ctx)               // [B][K][K][H]  (p = end-idx, q = start-idx)
{
    const int q = blockIdx.x;   // start-span index
    const int b = blockIdx.y;

    __shared__ int e_s[KK];     // end values, sorted descending
    __shared__ int p_s[KK];     // original span index of each sorted end

    if (threadIdx.x == 0) {
        int ev[KK], pi[KK];
        for (int p = 0; p < KK; ++p) {
            ev[p] = ids[((b * KK + p) * 2) + 1];
            pi[p] = p;
        }
        // selection sort, descending by end value (K=32, negligible cost)
        for (int a = 0; a < KK; ++a) {
            int best = a;
            for (int c = a + 1; c < KK; ++c)
                if (ev[c] > ev[best]) best = c;
            int te = ev[a]; ev[a] = ev[best]; ev[best] = te;
            int tp = pi[a]; pi[a] = pi[best]; pi[best] = tp;
            e_s[a] = ev[a];
            p_s[a] = pi[a];
        }
    }
    __syncthreads();

    const int hi = ids[((b * KK + q) * 2) + 0];  // start value of span q
    const int f  = threadIdx.x * 2;              // two features per thread

    const float* tok = tokens + (long)b * SS * HH;
    const unsigned char* mb = msk + (long)b * SS;

    float r0 = -INFINITY, r1 = -INFINITY;

    int ptr = 0;
    // ends >= hi give empty ranges (invalid pairs) -> never read; skip them
    while (ptr < KK && e_s[ptr] >= hi) ++ptr;

    for (int t = hi - 1; t >= 0 && ptr < KK; --t) {
        if (t >= 8)
            __builtin_prefetch(tok + (long)(t - 8) * HH + f, 0, 1);
        if (mb[t]) {
            r0 = fmaxf(r0, tok[(long)t * HH + f]);
            r1 = fmaxf(r1, tok[(long)t * HH + f + 1]);
        }
        while (ptr < KK && e_s[ptr] == t) {
            // running max now covers [t, hi) == [ends[p], starts[q])
            float* dst = ctx + (((long)(b * KK + p_s[ptr]) * KK + q) * HH) + f;
            dst[0] = r0;
            dst[1] = r1;
            ++ptr;
        }
    }
}

// ---------------------------------------------------------------------------
// rel-row construction: gather head / tail / context and emit bf16 rows.
// rel row layout: [head(512) | tail(512) | context(512)].
// ---------------------------------------------------------------------------
__global__ void __launch_bounds__(256)
build_rel(const float* __restrict__ spans,  // [B][K][H]
          const int* __restrict__ ids,      // [B][K][2]
          const float* __restrict__ ctx,    // [B][K][K][H]
          bf16_t* __restrict__ rel)         // [MROWS][H3]
{
    const int pairIdx = blockIdx.x;            // 0 .. B*K*K-1
    const int b  = pairIdx / (KK * KK);
    const int ij = pairIdx % (KK * KK);
    const int i  = ij / KK;
    const int j  = ij % KK;

    const int si = ids[((b * KK + i) * 2) + 0];
    const int ei = ids[((b * KK + i) * 2) + 1];
    const int sj = ids[((b * KK + j) * 2) + 0];
    const int ej = ids[((b * KK + j) * 2) + 1];
    const int lo = (ei < ej) ? ei : ej;        // min_end
    const int hi = (si > sj) ? si : sj;        // max_start
    const bool valid = lo < hi;
    const int p = (ei < ej) ? i : j;           // span giving min_end
    const int q = (si > sj) ? i : j;           // span giving max_start

    const int f = threadIdx.x * 2;

    const float* head = spans + (long)(b * KK + i) * HH;
    const float* tail = spans + (long)(b * KK + j) * HH;
    const float h0 = head[f], h1 = head[f + 1];
    const float t0 = tail[f], t1 = tail[f + 1];

    float c0, c1;
    if (valid) {
        const float* cp = ctx + (((long)(b * KK + p) * KK + q) * HH) + f;
        c0 = cp[0];
        c1 = cp[1];
    } else {
        c0 = h0;
        c1 = h1;
    }

    bf16_t* row = rel + (long)pairIdx * H3;
    row[f]              = (bf16_t)h0;
    row[f + 1]          = (bf16_t)h1;
    row[HH + f]         = (bf16_t)t0;
    row[HH + f + 1]     = (bf16_t)t1;
    row[2 * HH + f]     = (bf16_t)c0;
    row[2 * HH + f + 1] = (bf16_t)c1;
}

// ---------------------------------------------------------------------------
// WMMA GEMM:  out[M][N] = act( A[M][Kd] @ Bt[N][Kd]^T + bias )
//   A  : bf16 row-major [M][Kd];  Bt : bf16 row-major [N][Kd]
//   relu_bf16 = 1 -> ReLU, store bf16 ; 0 -> identity, store fp32
// Block = 256 threads = 8 waves; block tile 128x128; wave tile 32x64
//   = 2x4 v8f accumulators -> 8 v_wmma_f32_16x16x32_bf16 per K-step.
// ---------------------------------------------------------------------------
__global__ void __launch_bounds__(256)
gemm_wmma(const bf16_t* __restrict__ A,
          const bf16_t* __restrict__ Bt,
          const float*  __restrict__ bias,
          void* __restrict__ outp,
          int M, int N, int Kd, int relu_bf16)
{
    const int lane = threadIdx.x & 31;
    const int wave = threadIdx.x >> 5;
    const int wm   = wave & 3;                 // wave M index 0..3
    const int wn   = wave >> 2;                // wave N index 0..1
    const int m_base = blockIdx.y * 128 + wm * 32;
    const int n_base = blockIdx.x * 128 + wn * 64;

    const int half = lane >> 4;                // 0 or 1
    const int l16  = lane & 15;

    v8f acc[2][4];
#pragma unroll
    for (int mt = 0; mt < 2; ++mt)
#pragma unroll
        for (int nt = 0; nt < 4; ++nt)
#pragma unroll
            for (int r = 0; r < 8; ++r)
                acc[mt][nt][r] = 0.0f;

    for (int k0 = 0; k0 < Kd; k0 += 32) {
        v16bf afrag[2];
#pragma unroll
        for (int mt = 0; mt < 2; ++mt) {
            const bf16_t* ap =
                A + (long)(m_base + mt * 16 + l16) * Kd + k0 + half * 8;
            const v8bf alo = *(const v8bf*)(ap);        // K {0..7}/{8..15}
            const v8bf ahi = *(const v8bf*)(ap + 16);   // K {16..23}/{24..31}
#pragma unroll
            for (int e = 0; e < 8; ++e) {
                afrag[mt][e]     = alo[e];
                afrag[mt][e + 8] = ahi[e];
            }
        }

        v16bf bfrag[4];
#pragma unroll
        for (int nt = 0; nt < 4; ++nt) {
            const bf16_t* bp =
                Bt + (long)(n_base + nt * 16 + l16) * Kd + k0 + half * 8;
            const v8bf blo = *(const v8bf*)(bp);
            const v8bf bhi = *(const v8bf*)(bp + 16);
#pragma unroll
            for (int e = 0; e < 8; ++e) {
                bfrag[nt][e]     = blo[e];
                bfrag[nt][e + 8] = bhi[e];
            }
        }

#pragma unroll
        for (int mt = 0; mt < 2; ++mt)
#pragma unroll
            for (int nt = 0; nt < 4; ++nt)
                acc[mt][nt] = __builtin_amdgcn_wmma_f32_16x16x32_bf16(
                    /*neg_a=*/false, afrag[mt],
                    /*neg_b=*/false, bfrag[nt],
                    /*c_mod=*/(short)0, acc[mt][nt],
                    /*reuse_a=*/false, /*reuse_b=*/false);
    }

    // Epilogue: bias (+ReLU) and store.
    if (relu_bf16) {
        bf16_t* outb = (bf16_t*)outp;
#pragma unroll
        for (int mt = 0; mt < 2; ++mt) {
#pragma unroll
            for (int nt = 0; nt < 4; ++nt) {
                const int n  = n_base + nt * 16 + l16;
                const float bv = bias[n];
#pragma unroll
                for (int r = 0; r < 8; ++r) {
                    const int m = m_base + mt * 16 + half * 8 + r;
                    float v = acc[mt][nt][r] + bv;
                    v = fmaxf(v, 0.0f);
                    outb[(long)m * N + n] = (bf16_t)v;
                }
            }
        }
    } else {
        float* outf = (float*)outp;
#pragma unroll
        for (int mt = 0; mt < 2; ++mt) {
#pragma unroll
            for (int nt = 0; nt < 4; ++nt) {
                const int n  = n_base + nt * 16 + l16;
                const float bv = bias[n];
#pragma unroll
                for (int r = 0; r < 8; ++r) {
                    const int m = m_base + mt * 16 + half * 8 + r;
                    outf[(long)m * N + n] = acc[mt][nt][r] + bv;
                }
            }
        }
    }
}

// ---------------------------------------------------------------------------
// Launch
// ---------------------------------------------------------------------------
extern "C" void kernel_launch(void* const* d_in, const int* in_sizes, int n_in,
                              void* d_out, int out_size, void* d_ws, size_t ws_size,
                              hipStream_t stream) {
    const float*         spans  = (const float*)d_in[0];          // [B,K,H]
    const float*         tokens = (const float*)d_in[1];          // [B,S,H]
    const float*         W1     = (const float*)d_in[2];          // [3H,FFN]
    const float*         b1     = (const float*)d_in[3];          // [FFN]
    const float*         W2     = (const float*)d_in[4];          // [FFN,H]
    const float*         b2     = (const float*)d_in[5];          // [H]
    const int*           ids    = (const int*)d_in[6];            // [B,K,2]
    const unsigned char* masks  = (const unsigned char*)d_in[7];  // [B,S] bool
    float*               out    = (float*)d_out;                  // [B,K*K,H] f32

    // Workspace layout
    char* ws = (char*)d_ws;
    size_t off = 0;
    bf16_t* rel = (bf16_t*)(ws + off);                // [MROWS][H3]
    off += (size_t)MROWS * H3 * sizeof(bf16_t);       // 12,582,912
    bf16_t* W1t = (bf16_t*)(ws + off);                // [FFN][H3]
    off += (size_t)FFN * H3 * sizeof(bf16_t);         // 7,077,888
    bf16_t* hid = (bf16_t*)(ws + off);                // [MROWS][FFN]
    off += (size_t)MROWS * FFN * sizeof(bf16_t);      // 18,874,368
    bf16_t* W2t = (bf16_t*)(ws + off);                // [HH][FFN]
    off += (size_t)HH * FFN * sizeof(bf16_t);         // 2,359,296
    float*  ctx = (float*)(ws + off);                 // [B][K][K][H]
    off += (size_t)BB * KK * KK * HH * sizeof(float); // 8,388,608

    // 1) Weight transposes to bf16 (K-contiguous per output row)
    {
        long total1 = (long)H3 * FFN;
        transpose_to_bf16<<<(unsigned)((total1 + 255) / 256), 256, 0, stream>>>(
            W1, W1t, H3, FFN);
        long total2 = (long)FFN * HH;
        transpose_to_bf16<<<(unsigned)((total2 + 255) / 256), 256, 0, stream>>>(
            W2, W2t, FFN, HH);
    }

    // 2) Shared range-max table: one descending scan per (batch, start value)
    build_ctx<<<dim3(KK, BB), 256, 0, stream>>>(tokens, ids, masks, ctx);

    // 3) Gather head/tail/context into bf16 rel rows
    build_rel<<<MROWS, 256, 0, stream>>>(spans, ids, ctx, rel);

    // 4) hid = relu(rel @ W1 + b1)   [4096 x 2304], bf16 out
    {
        dim3 grid(FFN / 128, MROWS / 128);   // (18, 32)
        gemm_wmma<<<grid, 256, 0, stream>>>(rel, W1t, b1, (void*)hid,
                                            MROWS, FFN, H3, /*relu_bf16=*/1);
    }

    // 5) out = hid @ W2 + b2         [4096 x 512], fp32 out -> d_out
    {
        dim3 grid(HH / 128, MROWS / 128);    // (4, 32)
        gemm_wmma<<<grid, 256, 0, stream>>>(hid, W2t, b2, (void*)out,
                                            MROWS, HH, FFN, /*relu_bf16=*/0);
    }
}
